// TabNetModel_3693671875248
// MI455X (gfx1250) — matrix operand order repaired and checked by reading, not compiled, not run on hardware
//
#include <hip/hip_runtime.h>

typedef unsigned short u16;
typedef __attribute__((ext_vector_type(16))) __bf16 bf16x16;
typedef __attribute__((ext_vector_type(8)))  float  f32x8;

#define BATCH   32768
#define NFEAT   512
#define NDIM2   1024
#define ODIM    256
#define FCOEF   768
#define SQRTH   0.70710678118654752440f

__device__ __forceinline__ u16 f2bf(float f) {
  unsigned u = __float_as_uint(f);
  u = u + 0x7FFFu + ((u >> 16) & 1u);   // round-to-nearest-even
  return (u16)(u >> 16);
}

struct Frag32B { uint4 lo, hi; };
__device__ __forceinline__ bf16x16 make_frag(uint4 lo, uint4 hi) {
  Frag32B f; f.lo = lo; f.hi = hi;
  return __builtin_bit_cast(bf16x16, f);
}

// Flat->LDS aperture mapping uses addr[31:0] (ISA 10.2), and the async-LDS
// VDST operand is a byte offset in the workgroup LDS allocation.
__device__ __forceinline__ unsigned lds_off(const void* p) {
  return (unsigned)(unsigned long long)p;
}

// GLOBAL_LOAD_ASYNC_TO_LDS_B128: 16B/lane memory->LDS, tracked by ASYNCcnt,
// no VGPR round trip (CDNA5 ISA).
__device__ __forceinline__ void async_b128(unsigned ldsoff, const void* gptr) {
  asm volatile("global_load_async_to_lds_b128 %0, %1, off"
               :: "v"(ldsoff), "v"(gptr) : "memory");
}

// ---------------------------------------------------------------------------
// Tiled bf16 WMMA GEMM:  C[M,N] (f32) = A[M,K] @ Wt[N,K]^T
// AMODE 0: A bf16 row-major, staged via async-LDS copies.
// AMODE 1: A f32 row-major with fused batch-norm (x*scale[k]+shift[k]),
//          staged through VGPRs (loads batched, commit after WMMA block).
// 128x128x32 block tile, 8 waves (4x2), LDS double-buffered; async engine
// copies tile k+1 while WMMAs consume tile k. The s_wait_asynccnt is tied to
// the accumulators so the compiler cannot sink the WMMAs below the wait.
// ---------------------------------------------------------------------------
#define BM 128
#define BN 128
#define BK 32
#define KPAD 40        // padded LDS K-stride (breaks ds_read_b128 bank conflicts)
#define BUFB (BM * KPAD * 2)   // 10240 bytes per buffer (same for sA and sB)

template<int AMODE>
__global__ __launch_bounds__(256)
void gemm_wmma(const void* __restrict__ Aptr, const u16* __restrict__ Bt,
               float* __restrict__ C, int M, int N, int K,
               const float* __restrict__ scale, const float* __restrict__ shift)
{
  __shared__ u16 sA[2][BM * KPAD];
  __shared__ u16 sB[2][BN * KPAD];
  const int tid  = threadIdx.x;
  const int lane = tid & 31;
  const int wave = tid >> 5;
  const int wm   = wave >> 1;       // 0..3 -> 32-row strip
  const int wn   = wave & 1;        // 0..1 -> 64-col strip
  const int half = lane >> 4;       // WMMA lane group 0/1
  const int l16  = lane & 15;
  const size_t bm = (size_t)blockIdx.y * BM;
  const size_t bn = (size_t)blockIdx.x * BN;

  // ---- staging addresses (buffer 0; buffer 1 = +BUFB bytes) ---------------
  const u16* bsrc[2]; unsigned bdst[2];
  #pragma unroll
  for (int i = 0; i < 2; ++i) {
    int idx = tid + i * 256;            // 512 chunks of 8 halves (BN x BK)
    int r = idx >> 2, p = idx & 3;
    bsrc[i] = Bt + (bn + r) * (size_t)K + p * 8;
    bdst[i] = lds_off(&sB[0][r * KPAD + p * 8]);
  }
  const u16* asrc0[2]; unsigned adst0[2];
  if (AMODE == 0) {
    const u16* A = (const u16*)Aptr;
    #pragma unroll
    for (int i = 0; i < 2; ++i) {
      int idx = tid + i * 256;
      int r = idx >> 2, p = idx & 3;
      asrc0[i] = A + (bm + r) * (size_t)K + p * 8;
      adst0[i] = lds_off(&sA[0][r * KPAD + p * 8]);
    }
  }
  int a1p[4]; const float* a1src[4]; u16* a1dst[4];
  if (AMODE == 1) {
    const float* A = (const float*)Aptr;
    #pragma unroll
    for (int i = 0; i < 4; ++i) {
      int idx = tid + i * 256;          // 1024 chunks of 4 floats (BM x BK)
      int r = idx >> 3; a1p[i] = idx & 7;
      a1src[i] = A + (bm + r) * (size_t)K + a1p[i] * 4;
      a1dst[i] = &sA[0][r * KPAD + a1p[i] * 4];
    }
  }

  auto stage_async = [&](int k0, unsigned bufoff) {
    #pragma unroll
    for (int i = 0; i < 2; ++i) async_b128(bdst[i] + bufoff, bsrc[i] + k0);
    if (AMODE == 0) {
      #pragma unroll
      for (int i = 0; i < 2; ++i) async_b128(adst0[i] + bufoff, asrc0[i] + k0);
    }
  };
  auto a1_issue = [&](int k0, float4* regs) {
    #pragma unroll
    for (int i = 0; i < 4; ++i) regs[i] = *(const float4*)(a1src[i] + k0);
  };
  auto a1_commit = [&](int k0, unsigned bufoff, const float4* regs) {
    #pragma unroll
    for (int i = 0; i < 4; ++i) {
      int kc = k0 + a1p[i] * 4;
      u16 h0 = f2bf(regs[i].x * scale[kc + 0] + shift[kc + 0]);
      u16 h1 = f2bf(regs[i].y * scale[kc + 1] + shift[kc + 1]);
      u16 h2 = f2bf(regs[i].z * scale[kc + 2] + shift[kc + 2]);
      u16 h3 = f2bf(regs[i].w * scale[kc + 3] + shift[kc + 3]);
      uint2 pk;
      pk.x = (unsigned)h0 | ((unsigned)h1 << 16);
      pk.y = (unsigned)h2 | ((unsigned)h3 << 16);
      *(uint2*)((char*)a1dst[i] + bufoff) = pk;
    }
  };

  f32x8 acc[2][4] = {};

  // ---- prologue: stage tile 0 into buffer 0 -------------------------------
  {
    float4 areg[4];
    if (AMODE == 1) a1_issue(0, areg);
    stage_async(0, 0u);
    if (AMODE == 1) a1_commit(0, 0u, areg);
    asm volatile("s_wait_asynccnt 0x0" ::: "memory");
    __syncthreads();
  }

  unsigned curoff = 0;
  for (int k0 = 0; k0 < K; k0 += BK) {
    const u16* sAc = &sA[0][0] + (curoff >> 1);
    const u16* sBc = &sB[0][0] + (curoff >> 1);

    // fragment loads from current buffer.
    // A frag (16-bit 16x32): half 0 -> K{0..7,16..23}, half 1 -> K{8..15,24..31}
    // B frag (dense 32x16):  half 0 -> K{0..7,16..23}, half 1 -> K{8..15,24..31}
    bf16x16 afrag[2], bfrag[4];
    #pragma unroll
    for (int tm = 0; tm < 2; ++tm) {
      int row = wm * 32 + tm * 16 + l16;
      uint4 lo = *(const uint4*)&sAc[row * KPAD + half * 8];
      uint4 hi = *(const uint4*)&sAc[row * KPAD + 16 + half * 8];
      afrag[tm] = make_frag(lo, hi);
    }
    #pragma unroll
    for (int tn = 0; tn < 4; ++tn) {
      int nr = wn * 64 + tn * 16 + l16;
      uint4 lo = *(const uint4*)&sBc[nr * KPAD + half * 8];
      uint4 hi = *(const uint4*)&sBc[nr * KPAD + 16 + half * 8];
      bfrag[tn] = make_frag(lo, hi);
    }

    // overlap: async engine fills the other buffer while WMMAs run.
    // (safe: all waves' reads of the other buffer finished before the
    //  previous barrier, since frag ds_loads complete before their WMMAs.)
    const unsigned nxtoff = curoff ^ (unsigned)BUFB;
    const bool more = (k0 + BK) < K;
    float4 areg[4];
    if (more) {
      if (AMODE == 1) a1_issue(k0 + BK, areg);
      stage_async(k0 + BK, nxtoff);
    }

    #pragma unroll
    for (int tm = 0; tm < 2; ++tm)
      #pragma unroll
      for (int tn = 0; tn < 4; ++tn)
        acc[tm][tn] = __builtin_amdgcn_wmma_f32_16x16x32_bf16(
            false, afrag[tm], false, bfrag[tn], (short)0, acc[tm][tn], false, false);

    if (more && AMODE == 1) a1_commit(k0 + BK, nxtoff, areg);

    // Wait for tile k+1 to land -- tied to the accumulators so the WMMAs
    // cannot be sunk below this wait (keeps copy/compute overlap).
    asm volatile("s_wait_asynccnt 0x0"
                 : "+v"(acc[0][0]), "+v"(acc[0][1]), "+v"(acc[0][2]), "+v"(acc[0][3]),
                   "+v"(acc[1][0]), "+v"(acc[1][1]), "+v"(acc[1][2]), "+v"(acc[1][3])
                 :: "memory");
    __syncthreads();      // (also orders the AMODE1 ds_stores via dscnt)
    curoff = nxtoff;
  }

  // C layout: VGPR i -> row base+i (+8 for lanes 16..31), col = base + l16
  #pragma unroll
  for (int tm = 0; tm < 2; ++tm) {
    size_t rbase = bm + wm * 32 + tm * 16 + half * 8;
    #pragma unroll
    for (int tn = 0; tn < 4; ++tn) {
      size_t col = bn + wn * 64 + tn * 16 + l16;
      #pragma unroll
      for (int i = 0; i < 8; ++i)
        C[(rbase + i) * (size_t)N + col] = acc[tm][tn][i];
    }
  }
}

// ---------------------------------------------------------------------------
// Support kernels
// ---------------------------------------------------------------------------
__global__ void fill_f32(float* __restrict__ p, float v, size_t n) {
  size_t i = (size_t)blockIdx.x * blockDim.x + threadIdx.x;
  if (i < n) p[i] = v;
}

// W[K][N] f32 -> Wt[N][K] bf16
__global__ __launch_bounds__(256)
void transpose_bf16(const float* __restrict__ W, u16* __restrict__ Wt, int K, int N) {
  __shared__ float tile[32][33];
  int n0 = blockIdx.x * 32, k0 = blockIdx.y * 32;
  int tx = threadIdx.x & 31, ty = threadIdx.x >> 5;   // 32x8
  #pragma unroll
  for (int i = 0; i < 32; i += 8)
    tile[ty + i][tx] = W[(size_t)(k0 + ty + i) * N + n0 + tx];
  __syncthreads();
  #pragma unroll
  for (int i = 0; i < 32; i += 8)
    Wt[(size_t)(n0 + ty + i) * K + k0 + tx] = f2bf(tile[tx][ty + i]);
}

// column sum / sum-of-squares partials (sums[0..N)=sum, sums[N..2N)=sumsq)
#define CS_ROWS 1024
__global__ __launch_bounds__(256)
void colstats(const float* __restrict__ X, int N, float* __restrict__ sums) {
  int col = blockIdx.x * 256 + threadIdx.x;
  if (col >= N) return;
  size_t r0 = (size_t)blockIdx.y * CS_ROWS;
  float s = 0.f, ss = 0.f;
  for (size_t r = r0; r < r0 + CS_ROWS; ++r) {
    float v = X[r * (size_t)N + col];
    s += v; ss += v * v;
  }
  atomicAdd(&sums[col], s);
  atomicAdd(&sums[N + col], ss);
}

// mean/var -> affine (scale, shift) so bn(x) = x*scale + shift
__global__ void bn_finalize(const float* __restrict__ sums,
                            const float* __restrict__ g, const float* __restrict__ b,
                            int N, float invM,
                            float* __restrict__ scale, float* __restrict__ shift) {
  int c = blockIdx.x * 256 + threadIdx.x;
  if (c >= N) return;
  float mean = sums[c] * invM;
  float var  = sums[N + c] * invM - mean * mean;
  float sc   = g[c] * rsqrtf(var + 1e-5f);
  scale[c] = sc;
  shift[c] = b[c] - mean * sc;
}

// features = bn(data); masked(bf16) = features  (step-0 mask is identity)
__global__ void features_k(const float* __restrict__ data,
                           const float* __restrict__ scale, const float* __restrict__ shift,
                           float* __restrict__ features, u16* __restrict__ masked) {
  size_t i = (size_t)blockIdx.x * 256 + threadIdx.x;
  int c = (int)(i & (NFEAT - 1));
  float f = data[i] * scale[c] + shift[c];
  features[i] = f;
  masked[i] = f2bf(f);
}

// t2 = (glu_half(bn(g2)) + f1) * sqrt(1/2)  -> bf16
__global__ void glu_t2(const float* __restrict__ g2, const float* __restrict__ f1,
                       const float* __restrict__ scale, const float* __restrict__ shift,
                       u16* __restrict__ t2) {
  size_t i = (size_t)blockIdx.x * 256 + threadIdx.x;   // over BATCH*512
  size_t r = i >> 9; int j = (int)(i & 511);
  size_t base = r << 10;
  float a  = g2[base + j]       * scale[j]       + shift[j];
  float bb = g2[base + j + 512] * scale[j + 512] + shift[j + 512];
  float sig = 1.f / (1.f + expf(-bb));
  float ta = (a * sig + f1[base + j])       * SQRTH;
  float tb = (bb      + f1[base + j + 512]) * SQRTH;
  t2[base + j]       = f2bf(ta);
  t2[base + j + 512] = f2bf(tb);
}

// t4 epilogue: output_agg += relu(t4[:, :256]);  feats = t4[:, 256:] (bf16, 768 wide)
__global__ void glu_t4(const float* __restrict__ g4, const float* __restrict__ f3,
                       const float* __restrict__ scale, const float* __restrict__ shift,
                       float* __restrict__ out_agg, u16* __restrict__ feats) {
  size_t i = (size_t)blockIdx.x * 256 + threadIdx.x;   // over BATCH*512
  size_t r = i >> 9; int j = (int)(i & 511);
  size_t base = r << 10;
  float a  = g4[base + j]       * scale[j]       + shift[j];
  float bb = g4[base + j + 512] * scale[j + 512] + shift[j + 512];
  float sig = 1.f / (1.f + expf(-bb));
  float ta = (a * sig + f3[base + j])       * SQRTH;
  float tb = (bb      + f3[base + j + 512]) * SQRTH;
  if (j < ODIM) out_agg[r * ODIM + j] += fmaxf(ta, 0.f);
  else          feats[r * FCOEF + (j - ODIM)] = f2bf(ta);
  feats[r * FCOEF + j + 256] = f2bf(tb);   // cols 512..1023 -> feats 256..767
}

// sparsemax + prior/entropy update; one wave (32 lanes x 16 vals) per row.
// tau solves sum(max(z - tau, 0)) == 1 (monotone) -> binary search on
// [max(z)-1, max(z)]  ==  exact simplex projection, no sort needed.
__global__ __launch_bounds__(256)
void mask_k(const float* __restrict__ mgem,
            const float* __restrict__ scale, const float* __restrict__ shift,
            float* __restrict__ comp, const float* __restrict__ features,
            u16* __restrict__ masked, float* __restrict__ ent_out) {
  const int lane = threadIdx.x & 31;
  const size_t row = (size_t)blockIdx.x * 8 + (threadIdx.x >> 5);
  const size_t base = row * NFEAT;
  float z[16];
  float zmax = -3.0e38f;
  #pragma unroll
  for (int i = 0; i < 16; ++i) {
    int c = lane + i * 32;
    float v = (mgem[base + c] * scale[c] + shift[c]) * comp[base + c];
    z[i] = v;
    zmax = fmaxf(zmax, v);
  }
  #pragma unroll
  for (int m = 16; m >= 1; m >>= 1) zmax = fmaxf(zmax, __shfl_xor(zmax, m, 32));
  float lo = zmax - 1.f, hi = zmax;
  for (int it = 0; it < 40; ++it) {
    float tau = 0.5f * (lo + hi);
    float s = 0.f;
    #pragma unroll
    for (int i = 0; i < 16; ++i) s += fmaxf(z[i] - tau, 0.f);
    #pragma unroll
    for (int m = 16; m >= 1; m >>= 1) s += __shfl_xor(s, m, 32);
    if (s > 1.f) lo = tau; else hi = tau;
  }
  float tau = 0.5f * (lo + hi);
  float ent = 0.f;
  #pragma unroll
  for (int i = 0; i < 16; ++i) {
    int c = lane + i * 32;
    float mv = fmaxf(z[i] - tau, 0.f);
    comp[base + c] = comp[base + c] * (0.5f - mv);          // RELAX = 0.5
    ent += -mv * logf(mv + 1e-5f);
    masked[base + c] = f2bf(mv * features[base + c]);
  }
  #pragma unroll
  for (int m = 16; m >= 1; m >>= 1) ent += __shfl_xor(ent, m, 32);
  if (lane == 0) atomicAdd(ent_out, ent * (1.0f / (32768.0f * 5.0f)));
}

// ---------------------------------------------------------------------------
// Orchestration
// ---------------------------------------------------------------------------
extern "C" void kernel_launch(void* const* d_in, const int* in_sizes, int n_in,
                              void* d_out, int out_size, void* d_ws, size_t ws_size,
                              hipStream_t stream)
{
  (void)in_sizes; (void)n_in; (void)out_size; (void)ws_size;
  const float* data  = (const float*)d_in[0];
  const float* W1    = (const float*)d_in[1];
  const float* W2    = (const float*)d_in[2];
  const float* W3    = (const float*)d_in[3];
  const float* W4    = (const float*)d_in[4];
  const float* Wm    = (const float*)d_in[5];
  const float* bn_g  = (const float*)d_in[6];
  const float* bn_b  = (const float*)d_in[7];
  const float* bn1_g = (const float*)d_in[8];
  const float* bn1_b = (const float*)d_in[9];
  const float* bn2_g = (const float*)d_in[10];
  const float* bn2_b = (const float*)d_in[11];

  char* ws = (char*)d_ws; size_t off = 0;
  auto alloc = [&](size_t bytes) -> void* {
    void* p = ws + off; off += (bytes + 255) & ~(size_t)255; return p;
  };
  float* features = (float*)alloc((size_t)BATCH * NFEAT * 4);
  float* comp     = (float*)alloc((size_t)BATCH * NFEAT * 4);
  u16*   maskedbf = (u16*)  alloc((size_t)BATCH * NFEAT * 2);
  float* actA     = (float*)alloc((size_t)BATCH * NDIM2 * 4);  // f1/f3/mgem
  float* actB     = (float*)alloc((size_t)BATCH * NDIM2 * 4);  // g2/g4
  u16*   t2feats  = (u16*)  alloc((size_t)BATCH * NDIM2 * 2);  // t2 & feats
  u16* W1t = (u16*)alloc((size_t)512 * 1024 * 2);
  u16* W2t = (u16*)alloc((size_t)1024 * 1024 * 2);
  u16* W3t = (u16*)alloc((size_t)1024 * 1024 * 2);
  u16* W4t = (u16*)alloc((size_t)1024 * 1024 * 2);
  u16* Wmt = (u16*)alloc((size_t)768 * 512 * 2);
  float* sums = (float*)alloc(2 * 1024 * 4);
  float* scl  = (float*)alloc(1024 * 4);
  float* shf  = (float*)alloc(1024 * 4);

  // zero output (output_agg accumulates; last element = entropy scalar)
  size_t nout = (size_t)BATCH * ODIM + 1;
  fill_f32<<<dim3((unsigned)((nout + 255) / 256)), 256, 0, stream>>>((float*)d_out, 0.f, nout);

  // one-time weight prep: f32 [K][N] -> bf16 [N][K]
  transpose_bf16<<<dim3(1024 / 32, 512 / 32),  256, 0, stream>>>(W1, W1t, 512, 1024);
  transpose_bf16<<<dim3(1024 / 32, 1024 / 32), 256, 0, stream>>>(W2, W2t, 1024, 1024);
  transpose_bf16<<<dim3(1024 / 32, 1024 / 32), 256, 0, stream>>>(W3, W3t, 1024, 1024);
  transpose_bf16<<<dim3(1024 / 32, 1024 / 32), 256, 0, stream>>>(W4, W4t, 1024, 1024);
  transpose_bf16<<<dim3(512 / 32, 768 / 32),   256, 0, stream>>>(Wm, Wmt, 768, 512);

  auto run_stats = [&](const float* X, int N, const float* g, const float* b) {
    fill_f32<<<dim3((unsigned)((2 * N + 255) / 256)), 256, 0, stream>>>(sums, 0.f, (size_t)2 * N);
    colstats<<<dim3(N / 256, BATCH / CS_ROWS), 256, 0, stream>>>(X, N, sums);
    bn_finalize<<<dim3((N + 255) / 256), 256, 0, stream>>>(sums, g, b, N, 1.0f / (float)BATCH, scl, shf);
  };

  // features = bn(data); masked = features; comp = 1
  run_stats(data, NFEAT, bn_g, bn_b);
  features_k<<<dim3(BATCH * NFEAT / 256), 256, 0, stream>>>(data, scl, shf, features, maskedbf);
  fill_f32<<<dim3(BATCH * NFEAT / 256), 256, 0, stream>>>(comp, 1.0f, (size_t)BATCH * NFEAT);

  const dim3 gBig(NDIM2 / BN, BATCH / BM);   // N=1024 GEMMs
  const dim3 gMask(NFEAT / BN, BATCH / BM);  // N=512 mask GEMM

  for (int ni = 1; ni <= 5; ++ni) {
    // f1 = masked @ W1
    gemm_wmma<0><<<gBig, 256, 0, stream>>>(maskedbf, W1t, actA, BATCH, NDIM2, NFEAT, nullptr, nullptr);
    run_stats(actA, NDIM2, bn1_g, bn1_b);
    // g2 = bn(f1) @ W2   (BN fused into A-tile staging)
    gemm_wmma<1><<<gBig, 256, 0, stream>>>(actA, W2t, actB, BATCH, NDIM2, NDIM2, scl, shf);
    run_stats(actB, NDIM2, bn1_g, bn1_b);
    // t2 = (glu(bn(g2)) + f1) * sqrt(.5)
    glu_t2<<<dim3(BATCH * 512 / 256), 256, 0, stream>>>(actB, actA, scl, shf, t2feats);
    // f3 = t2 @ W3
    gemm_wmma<0><<<gBig, 256, 0, stream>>>(t2feats, W3t, actA, BATCH, NDIM2, NDIM2, nullptr, nullptr);
    run_stats(actA, NDIM2, bn1_g, bn1_b);
    // g4 = bn(f3) @ W4
    gemm_wmma<1><<<gBig, 256, 0, stream>>>(actA, W4t, actB, BATCH, NDIM2, NDIM2, scl, shf);
    run_stats(actB, NDIM2, bn1_g, bn1_b);
    // t4 epilogue: out_agg += relu(.), feats = t4[:,256:]
    glu_t4<<<dim3(BATCH * 512 / 256), 256, 0, stream>>>(actB, actA, scl, shf, (float*)d_out, t2feats);
    if (ni < 5) {
      // mgem = feats @ Wm  (reuses actA)
      gemm_wmma<0><<<gMask, 256, 0, stream>>>(t2feats, Wmt, actA, BATCH, NFEAT, FCOEF, nullptr, nullptr);
      run_stats(actA, NFEAT, bn2_g, bn2_b);
      mask_k<<<dim3(BATCH / 8), 256, 0, stream>>>(actA, scl, shf, comp, features, maskedbf,
                                                  ((float*)d_out) + (size_t)BATCH * ODIM);
    }
  }
}
